// GraphRNADISTAtteNCionE_79998060855522
// MI455X (gfx1250) — compile-verified, matrix operand order checked by reading, not run on hardware
//
#include <hip/hip_runtime.h>
#include <hip/hip_bf16.h>
#include <stdint.h>

#define BB 2
#define LL 192
#define EE 64
#define HH 4
#define CC 32
#define HC 128
#define MM (BB*LL*LL)   // 73728 pair positions

typedef __bf16 bf16_t;
typedef __attribute__((ext_vector_type(16))) __bf16 bfrag;
typedef __attribute__((ext_vector_type(8)))  float  ffrag;
typedef __attribute__((ext_vector_type(4)))  unsigned int u32x4;

union fragU { u32x4 v[2]; bfrag f; };

static __device__ __forceinline__ ffrag wmma_bf16(bfrag a, bfrag b, ffrag c){
  return __builtin_amdgcn_wmma_f32_16x16x32_bf16(false, a, false, b, (short)0, c, false, false);
}

// One fragment = two contiguous 16B runs per lane -> 2x ds_load_b128.
// Works for A tiles (row = M) and for B tiles stored transposed [N][K] (row = N).
static __device__ __forceinline__ bfrag load_frag(const bf16_t* base, int stride){
  int l = threadIdx.x & 31;
  const bf16_t* r = base + (size_t)(l & 15)*stride + (l >> 4)*8;
  fragU u;
  u.v[0] = *(const u32x4*)r;
  u.v[1] = *(const u32x4*)(r + 16);
  return u.f;
}

// ---- CDNA5 async global->LDS (ASYNCcnt-tracked, no VGPR staging) ----
static __device__ __forceinline__ uint32_t lds_off(const void* p){
  return (uint32_t)(size_t)p;            // LDS aperture: addr[31:0] = LDS offset
}
static __device__ __forceinline__ void async_load_b128(uint32_t dst_lds, const void* gsrc){
  asm volatile("global_load_async_to_lds_b128 %0, %1, off"
               :: "v"(dst_lds), "v"((unsigned long long)(size_t)gsrc)
               : "memory");
}
static __device__ __forceinline__ void async_wait0(){
  asm volatile("s_wait_asynccnt 0x0" ::: "memory");
}

// ---------------- weight prep (all transposed to [N][K] for B-fragments) ----------------
// out [512][64]: row n = output column (q0-127,k128-255,v256-383,fu384-511), col = E
__global__ void pack_qkvfu_t_kernel(const float* wq, const float* wk, const float* wv,
                                    const float* wfu, bf16_t* out){
  int idx = blockIdx.x*blockDim.x + threadIdx.x;
  if(idx >= 512*EE) return;
  int c = idx >> 6, e = idx & 63;
  const float* src = (c < 128) ? wq : (c < 256) ? wk : (c < 384) ? wv : wfu;
  out[idx] = (bf16_t)src[e*HC + (c & 127)];
}

// wo [128][64] -> out [64][128] (row n, col k)
__global__ void wo_t_kernel(const float* wo, bf16_t* out){
  int idx = blockIdx.x*blockDim.x + threadIdx.x;
  if(idx >= EE*HC) return;
  int n = idx >> 7, k = idx & 127;
  out[idx] = (bf16_t)wo[k*EE + n];
}

// w [O][I][5][5] -> out [25][O][I]  (per-tap, row n = O, col k = I)
__global__ void conv_w_perm_kernel(const float* w, bf16_t* out, int O, int I){
  int idx = blockIdx.x*blockDim.x + threadIdx.x;
  if(idx >= O*I*25) return;
  int t = idx / (I*O); int rem = idx % (I*O);
  int o = rem / I;     int i  = rem % I;
  out[idx] = (bf16_t)w[((size_t)(o*I + i))*25 + t];
}

// ---------------- layernorm (with optional spatial transpose) ----------------
__global__ void ln_kernel(const float* x, const float* g, const float* b,
                          bf16_t* xn, int tmode){
  int row = blockIdx.x*blockDim.x + threadIdx.x;
  if(row >= MM) return;
  int n = row / (LL*LL); int rem = row % (LL*LL);
  int a = rem / LL, bb = rem % LL;
  int gi = tmode ? bb : a, gj = tmode ? a : bb;
  const float* src = x + ((size_t)(n*LL + gi)*LL + gj)*EE;
  float mu = 0.f;
  #pragma unroll 8
  for(int e=0; e<EE; ++e) mu += src[e];
  mu *= (1.f/EE);
  float var = 0.f;
  #pragma unroll 8
  for(int e=0; e<EE; ++e){ float d = src[e]-mu; var += d*d; }
  float inv = rsqrtf(var*(1.f/EE) + 1e-5f);
  bf16_t* dst = xn + (size_t)row*EE;
  #pragma unroll 8
  for(int e=0; e<EE; ++e) dst[e] = (bf16_t)((src[e]-mu)*inv*g[e] + b[e]);
}

// ---------------- fused q/k/v/fu projection: M x 64 x 512 ----------------
__global__ void gemm_qkvfu_kernel(const bf16_t* xn, const bf16_t* wt, bf16_t* out){
  extern __shared__ char smem[];
  bf16_t* sA = (bf16_t*)smem;            // [128][64]
  bf16_t* sW = sA + 128*EE;              // [512][64] transposed weights
  int tid = threadIdx.x;
  size_t m0 = (size_t)blockIdx.x * 128;
  uint32_t sa_off = lds_off(sA), sw_off = lds_off(sW);
  const char* ga = (const char*)(xn + m0*EE);        // 16 KB
  #pragma unroll
  for(int i=0;i<4;++i)  async_load_b128(sa_off + tid*16 + i*4096, ga + tid*16 + i*4096);
  const char* gw = (const char*)wt;                  // 64 KB
  #pragma unroll
  for(int i=0;i<16;++i) async_load_b128(sw_off + tid*16 + i*4096, gw + tid*16 + i*4096);
  async_wait0();
  __syncthreads();
  int wave = tid >> 5;
  bfrag a0 = load_frag(sA + wave*16*EE,      EE);
  bfrag a1 = load_frag(sA + wave*16*EE + 32, EE);
  int l = tid & 31, colb = l & 15, rowadd = 8*(l >> 4);
  for(int nt=0; nt<32; ++nt){
    ffrag acc = {};
    bfrag b0 = load_frag(sW + (size_t)nt*16*EE,      EE);
    bfrag b1 = load_frag(sW + (size_t)nt*16*EE + 32, EE);
    acc = wmma_bf16(a0, b0, acc);
    acc = wmma_bf16(a1, b1, acc);
    float s = (nt < 8) ? 0.17677669529663687f : 1.f;   // q * C^-0.5
    int col = nt*16 + colb;
    #pragma unroll
    for(int r=0;r<8;++r){
      int m = wave*16 + r + rowadd;
      out[(m0+m)*512 + col] = (bf16_t)(acc[r]*s);
    }
  }
}

// ---------------- pair bias: pb = xn @ wb  (M x 64 x 4) ----------------
__global__ void pb_kernel(const bf16_t* xn, const float* wb, float* pb){
  int idx = blockIdx.x*blockDim.x + threadIdx.x;
  if(idx >= MM*HH) return;
  int row = idx >> 2, h = idx & 3;
  const bf16_t* r = xn + (size_t)row*EE;
  float s = 0.f;
  #pragma unroll 8
  for(int e=0; e<EE; ++e) s += (float)r[e] * wb[e*HH + h];
  pb[idx] = s;
}

// ---------------- triangular attention core: one block per (n,s,h) ----------------
__global__ void attn_kernel(const bf16_t* qkvfu, const float* pb, const float* mask,
                            bf16_t* outO, int tmode){
  int bid = blockIdx.x;
  int h = bid & 3; int t = bid >> 2;
  int s = t % LL, n = t / LL;
  extern __shared__ char smem[];
  bf16_t* sQ  = (bf16_t*)smem;          // [192][32]
  bf16_t* sK  = sQ + LL*CC;             // [192][32]
  bf16_t* sVT = sK + LL*CC;             // [32][192]  (V transposed for B-fragments)
  bf16_t* sP  = sVT + LL*CC;            // [192][192]
  int tid = threadIdx.x;                // 384
  size_t rowbase = ((size_t)(n*LL + s))*LL*512;
  // async-stage Q and K rows (64 B per row = 4 aligned b128 chunks)
  uint32_t sq_off = lds_off(sQ), sk_off = lds_off(sK);
  const char* basec = (const char*)(qkvfu + rowbase) + h*CC*2;
  for(int c2=tid; c2<LL*4; c2+=384){
    int r = c2 >> 2, part = (c2 & 3)*16;
    async_load_b128(sq_off + r*64 + part, basec + (size_t)r*1024 +       part);
    async_load_b128(sk_off + r*64 + part, basec + (size_t)r*1024 + 256 + part);
  }
  // V transposed manually
  for(int i2=tid; i2<LL*CC; i2+=384){
    int r = i2 >> 5, c = i2 & 31;
    sVT[(size_t)c*LL + r] = qkvfu[rowbase + (size_t)r*512 + 256 + h*CC + c];
  }
  async_wait0();
  __syncthreads();
  int wave = tid >> 5;                   // 0..11, strip rows [q0, q0+16)
  int q0 = wave*16;
  int l = tid & 31, colb = l & 15, rowadd = 8*(l >> 4);

  // scores: S[q, v] = sum_c K[q,c]*Q[v,c]  (keys-first einsum)
  ffrag Sf[12];
  bfrag ak = load_frag(sK + (size_t)q0*CC, CC);
  #pragma unroll
  for(int vt=0; vt<12; ++vt){
    ffrag z = {};
    bfrag bq = load_frag(sQ + (size_t)vt*16*CC, CC);  // B[c][v] from row-major Q
    Sf[vt] = wmma_bf16(ak, bq, z);
  }
  // + mask bias + pair bias
  #pragma unroll
  for(int vt=0; vt<12; ++vt){
    #pragma unroll
    for(int r=0;r<8;++r){
      int v = vt*16 + colb, q = q0 + r + rowadd;
      int mi = tmode ? v : q, mj = tmode ? q : v;
      float bias = 1e9f*(mask[((size_t)n*LL + mi)*LL + mj] - 1.f);
      float pbv  = pb[(((size_t)(n*LL + q))*LL + v)*HH + h];
      Sf[vt][r] += bias + pbv;
    }
  }
  // softmax over v: 16-lane reductions
  float rs[8];
  #pragma unroll
  for(int r=0;r<8;++r){
    float mx = -3.0e38f;
    #pragma unroll
    for(int vt=0; vt<12; ++vt) mx = fmaxf(mx, Sf[vt][r]);
    mx = fmaxf(mx, __shfl_xor(mx, 1, 16));
    mx = fmaxf(mx, __shfl_xor(mx, 2, 16));
    mx = fmaxf(mx, __shfl_xor(mx, 4, 16));
    mx = fmaxf(mx, __shfl_xor(mx, 8, 16));
    float sum = 0.f;
    #pragma unroll
    for(int vt=0; vt<12; ++vt){ Sf[vt][r] = __expf(Sf[vt][r]-mx); sum += Sf[vt][r]; }
    sum += __shfl_xor(sum, 1, 16);
    sum += __shfl_xor(sum, 2, 16);
    sum += __shfl_xor(sum, 4, 16);
    sum += __shfl_xor(sum, 8, 16);
    rs[r] = 1.f/sum;
  }
  #pragma unroll
  for(int vt=0; vt<12; ++vt)
    #pragma unroll
    for(int r=0;r<8;++r){
      int q = q0 + r + rowadd, v = vt*16 + colb;
      sP[(size_t)q*LL + v] = (bf16_t)(Sf[vt][r]*rs[r]);   // wave-private strip
    }
  // out = P @ V  (16 x 192 x 32), gated by silu(fu)
  #pragma unroll
  for(int nt=0; nt<2; ++nt){
    ffrag o = {};
    #pragma unroll
    for(int kc=0; kc<6; ++kc){
      bfrag ap = load_frag(sP  + (size_t)q0*LL     + kc*32, LL);
      bfrag bv = load_frag(sVT + (size_t)nt*16*LL  + kc*32, LL);
      o = wmma_bf16(ap, bv, o);
    }
    #pragma unroll
    for(int r=0;r<8;++r){
      int q = q0 + r + rowadd, c = nt*16 + colb;
      float fu = (float)qkvfu[rowbase + (size_t)q*512 + 384 + h*CC + c];
      float gate = fu / (1.f + __expf(-fu));
      outO[(((size_t)(n*LL + s))*LL + q)*HC + h*CC + c] = (bf16_t)(o[r]*gate);
    }
  }
}

// ---------------- output projection + mask + residual: M x 128 x 64 ----------------
__global__ void outproj_kernel(const bf16_t* O, const bf16_t* wot, const float* bo,
                               const float* mask, const float* xin,
                               float* outPR, bf16_t* outBF, int tmode){
  extern __shared__ char smem[];
  bf16_t* sA = (bf16_t*)smem;            // [128][128]
  bf16_t* sW = sA + 128*HC;              // [64][128] transposed wo
  int tid = threadIdx.x;
  size_t m0 = (size_t)blockIdx.x * 128;
  uint32_t sa_off = lds_off(sA), sw_off = lds_off(sW);
  const char* ga = (const char*)(O + m0*HC);         // 32 KB
  #pragma unroll
  for(int i=0;i<8;++i) async_load_b128(sa_off + tid*16 + i*4096, ga + tid*16 + i*4096);
  const char* gw = (const char*)wot;                 // 16 KB
  #pragma unroll
  for(int i=0;i<4;++i) async_load_b128(sw_off + tid*16 + i*4096, gw + tid*16 + i*4096);
  async_wait0();
  __syncthreads();
  int wave = tid >> 5;
  int l = tid & 31, colb = l & 15, rowadd = 8*(l >> 4);
  for(int nt=0; nt<4; ++nt){
    ffrag acc = {};
    #pragma unroll
    for(int ks=0; ks<4; ++ks){
      bfrag a = load_frag(sA + (size_t)wave*16*HC + ks*32, HC);
      bfrag b = load_frag(sW + (size_t)nt*16*HC   + ks*32, HC);
      acc = wmma_bf16(a, b, acc);
    }
    int col = nt*16 + colb;
    #pragma unroll
    for(int r=0;r<8;++r){
      size_t m = m0 + wave*16 + r + rowadd;
      int n = (int)(m / (LL*LL)); int rem = (int)(m % (LL*LL));
      int a_ = rem / LL, b_ = rem % LL;          // local (s, q)
      float mk = mask[((size_t)n*LL + a_)*LL + b_];
      int gi = tmode ? b_ : a_, gj = tmode ? a_ : b_;
      size_t gidx = (((size_t)(n*LL + gi))*LL + gj)*EE + col;
      float y = (acc[r] + bo[col])*mk + xin[gidx];
      outPR[gidx] = y;
      if(outBF) outBF[gidx] = (bf16_t)y;
    }
  }
}

// ---------------- conv1: 64 -> 128 channels, 5x5, silu, implicit GEMM ----------------
__global__ void conv1_kernel(const bf16_t* in, const bf16_t* wperm, const float* cb,
                             bf16_t* outT){
  extern __shared__ char smem[];
  bf16_t* sIn = (bf16_t*)smem;           // [5][68][64]
  bf16_t* sW  = sIn + 5*68*EE;           // [128][64] per-tap transposed slice
  int tid = threadIdx.x;
  int n = blockIdx.x / (LL*3); int rem = blockIdx.x % (LL*3);
  int i = rem / 3, j0 = (rem % 3)*64;
  uint32_t sin_off = lds_off(sIn), sw_off = lds_off(sW);
  // input halo tile: 16B chunks, async where in-range, LDS zero-fill where OOB
  for(int ch=tid; ch<5*68*8; ch+=256){   // 8 chunks of 16B per pixel (64ch * 2B)
    int r = ch >> 3; int part = (ch & 7)*16;
    int jl = r % 68, dy = r / 68;
    int gi = i + dy - 2, gj = j0 + jl - 2;
    if(gi>=0 && gi<LL && gj>=0 && gj<LL){
      const char* src = (const char*)(in + (((size_t)(n*LL + gi))*LL + gj)*EE) + part;
      async_load_b128(sin_off + r*128 + part, src);
    } else {
      u32x4 z = {};
      *((u32x4*)((char*)sIn + r*128 + part)) = z;
    }
  }
  async_wait0();
  ffrag acc[4];
  #pragma unroll
  for(int mt=0; mt<4; ++mt)
    #pragma unroll
    for(int r=0;r<8;++r) acc[mt][r] = 0.f;
  int wave = tid >> 5;                    // n-tile of 16 output channels
  for(int t=0; t<25; ++t){
    __syncthreads();
    const char* wsrc = (const char*)(wperm + (size_t)t*HC*EE);   // 16 KB slice
    #pragma unroll
    for(int it=0; it<4; ++it)
      async_load_b128(sw_off + tid*16 + it*4096, wsrc + tid*16 + it*4096);
    async_wait0();
    __syncthreads();
    int dy = t/5, dx = t%5;
    #pragma unroll
    for(int ks=0; ks<2; ++ks){
      bfrag b = load_frag(sW + (size_t)wave*16*EE + ks*32, EE);
      #pragma unroll
      for(int mt=0; mt<4; ++mt){
        bfrag a = load_frag(sIn + ((size_t)(dy*68 + mt*16 + dx))*EE + ks*32, EE);
        acc[mt] = wmma_bf16(a, b, acc[mt]);
      }
    }
  }
  int l = tid & 31, colb = l & 15, rowadd = 8*(l >> 4);
  int o = wave*16 + colb;
  float bias = cb[o];
  #pragma unroll
  for(int mt=0; mt<4; ++mt)
    #pragma unroll
    for(int r=0;r<8;++r){
      int p = mt*16 + r + rowadd;
      float y = acc[mt][r] + bias;
      y = y / (1.f + __expf(-y));                       // silu
      outT[(((size_t)(n*LL + i))*LL + j0 + p)*HC + o] = (bf16_t)y;
    }
}

// ---------------- conv2: 128 -> 64 channels, 5x5, + bias + residual, * mask ----------------
__global__ void conv2_kernel(const bf16_t* in, const bf16_t* wperm, const float* cb,
                             const float* resid, const float* mask, float* outp){
  extern __shared__ char smem[];
  bf16_t* sIn = (bf16_t*)smem;           // [5][68][128]
  bf16_t* sW  = sIn + 5*68*HC;           // [64][128] per-tap transposed slice
  int tid = threadIdx.x;
  int n = blockIdx.x / (LL*3); int rem = blockIdx.x % (LL*3);
  int i = rem / 3, j0 = (rem % 3)*64;
  uint32_t sin_off = lds_off(sIn), sw_off = lds_off(sW);
  for(int ch=tid; ch<5*68*16; ch+=256){  // 16 chunks of 16B per pixel (128ch * 2B)
    int r = ch >> 4; int part = (ch & 15)*16;
    int jl = r % 68, dy = r / 68;
    int gi = i + dy - 2, gj = j0 + jl - 2;
    if(gi>=0 && gi<LL && gj>=0 && gj<LL){
      const char* src = (const char*)(in + (((size_t)(n*LL + gi))*LL + gj)*HC) + part;
      async_load_b128(sin_off + r*256 + part, src);
    } else {
      u32x4 z = {};
      *((u32x4*)((char*)sIn + r*256 + part)) = z;
    }
  }
  async_wait0();
  ffrag acc[2];
  #pragma unroll
  for(int mi=0; mi<2; ++mi)
    #pragma unroll
    for(int r=0;r<8;++r) acc[mi][r] = 0.f;
  int wave = tid >> 5;
  int ntile = wave & 3, mhalf = wave >> 2;
  for(int t=0; t<25; ++t){
    __syncthreads();
    const char* wsrc = (const char*)(wperm + (size_t)t*HC*EE);   // 16 KB slice
    #pragma unroll
    for(int it=0; it<4; ++it)
      async_load_b128(sw_off + tid*16 + it*4096, wsrc + tid*16 + it*4096);
    async_wait0();
    __syncthreads();
    int dy = t/5, dx = t%5;
    #pragma unroll
    for(int ks=0; ks<4; ++ks){
      bfrag b = load_frag(sW + (size_t)ntile*16*HC + ks*32, HC);
      #pragma unroll
      for(int mi=0; mi<2; ++mi){
        int mt = mhalf*2 + mi;
        bfrag a = load_frag(sIn + ((size_t)(dy*68 + mt*16 + dx))*HC + ks*32, HC);
        acc[mi] = wmma_bf16(a, b, acc[mi]);
      }
    }
  }
  int l = tid & 31, colb = l & 15, rowadd = 8*(l >> 4);
  int o = ntile*16 + colb;
  #pragma unroll
  for(int mi=0; mi<2; ++mi)
    #pragma unroll
    for(int r=0;r<8;++r){
      int p = (mhalf*2 + mi)*16 + r + rowadd;
      size_t pix = ((size_t)(n*LL + i))*LL + j0 + p;
      float y = acc[mi][r] + cb[o] + resid[pix*EE + o];
      outp[pix*EE + o] = y * mask[pix];
    }
}

// ---------------- host ----------------
extern "C" void kernel_launch(void* const* d_in, const int* in_sizes, int n_in,
                              void* d_out, int out_size, void* d_ws, size_t ws_size,
                              hipStream_t stream) {
  const float* pair   = (const float*)d_in[0];
  const float* mask   = (const float*)d_in[1];
  const float* ain_g  = (const float*)d_in[2];
  const float* ain_b  = (const float*)d_in[3];
  const float* ain_wq = (const float*)d_in[4];
  const float* ain_wk = (const float*)d_in[5];
  const float* ain_wv = (const float*)d_in[6];
  const float* ain_wb = (const float*)d_in[7];
  const float* ain_wfu= (const float*)d_in[8];
  const float* ain_wo = (const float*)d_in[9];
  const float* ain_bo = (const float*)d_in[11];
  const float* aout_g  = (const float*)d_in[12];
  const float* aout_b  = (const float*)d_in[13];
  const float* aout_wq = (const float*)d_in[14];
  const float* aout_wk = (const float*)d_in[15];
  const float* aout_wv = (const float*)d_in[16];
  const float* aout_wb = (const float*)d_in[17];
  const float* aout_wfu= (const float*)d_in[18];
  const float* aout_wo = (const float*)d_in[19];
  const float* aout_bo = (const float*)d_in[21];
  const float* c1w = (const float*)d_in[22];
  const float* c1b = (const float*)d_in[23];
  const float* c2w = (const float*)d_in[24];
  const float* c2b = (const float*)d_in[25];
  float* out = (float*)d_out;

  char* ws = (char*)d_ws; size_t off = 0;
  auto alloc = [&](size_t bytes)->void*{
    void* p = ws + off; off += (bytes + 255) & ~(size_t)255; return p;
  };
  bf16_t* WQKVFU_A = (bf16_t*)alloc((size_t)512*EE*2);
  bf16_t* WQKVFU_B = (bf16_t*)alloc((size_t)512*EE*2);
  bf16_t* WO_A     = (bf16_t*)alloc((size_t)EE*HC*2);
  bf16_t* WO_B     = (bf16_t*)alloc((size_t)EE*HC*2);
  bf16_t* CW1      = (bf16_t*)alloc((size_t)25*HC*EE*2);
  bf16_t* CW2      = (bf16_t*)alloc((size_t)25*EE*HC*2);
  bf16_t* XN       = (bf16_t*)alloc((size_t)MM*EE*2);
  bf16_t* QKVFU    = (bf16_t*)alloc((size_t)MM*512*2);
  float*  PBUF     = (float*) alloc((size_t)MM*HH*4);
  bf16_t* OBUF     = (bf16_t*)alloc((size_t)MM*HC*2);
  float*  PR1      = (float*) alloc((size_t)MM*EE*4);
  float*  PR2      = (float*) alloc((size_t)MM*EE*4);
  bf16_t* PR2BF    = (bf16_t*)alloc((size_t)MM*EE*2);
  bf16_t* T1       = QKVFU;   // alias: QKVFU dead once convs start

  const int SH_GEMM = (128*EE + 512*EE)*2;          // 81920
  const int SH_ATTN = (3*LL*CC + LL*LL)*2;          // 110592
  const int SH_PROJ = (128*HC + EE*HC)*2;           // 49152
  const int SH_C1   = (5*68*EE + HC*EE)*2;          // 59904
  const int SH_C2   = (5*68*HC + EE*HC)*2;          // 103424

  // ---- weight prep ----
  pack_qkvfu_t_kernel<<<(512*EE+255)/256, 256, 0, stream>>>(ain_wq, ain_wk, ain_wv, ain_wfu, WQKVFU_A);
  pack_qkvfu_t_kernel<<<(512*EE+255)/256, 256, 0, stream>>>(aout_wq, aout_wk, aout_wv, aout_wfu, WQKVFU_B);
  wo_t_kernel<<<(EE*HC+255)/256, 256, 0, stream>>>(ain_wo,  WO_A);
  wo_t_kernel<<<(EE*HC+255)/256, 256, 0, stream>>>(aout_wo, WO_B);
  conv_w_perm_kernel<<<(25*HC*EE+255)/256, 256, 0, stream>>>(c1w, CW1, HC, EE);
  conv_w_perm_kernel<<<(25*EE*HC+255)/256, 256, 0, stream>>>(c2w, CW2, EE, HC);

  // ---- stage A: triangular attention, mode 'in' ----
  ln_kernel<<<(MM+255)/256, 256, 0, stream>>>(pair, ain_g, ain_b, XN, 0);
  gemm_qkvfu_kernel<<<MM/128, 256, SH_GEMM, stream>>>(XN, WQKVFU_A, QKVFU);
  pb_kernel<<<(MM*HH+255)/256, 256, 0, stream>>>(XN, ain_wb, PBUF);
  attn_kernel<<<BB*LL*HH, 384, SH_ATTN, stream>>>(QKVFU, PBUF, mask, OBUF, 0);
  outproj_kernel<<<MM/128, 256, SH_PROJ, stream>>>(OBUF, WO_A, ain_bo, mask, pair, PR1, nullptr, 0);

  // ---- stage B: triangular attention, mode 'out' ----
  ln_kernel<<<(MM+255)/256, 256, 0, stream>>>(PR1, aout_g, aout_b, XN, 1);
  gemm_qkvfu_kernel<<<MM/128, 256, SH_GEMM, stream>>>(XN, WQKVFU_B, QKVFU);
  pb_kernel<<<(MM*HH+255)/256, 256, 0, stream>>>(XN, aout_wb, PBUF);
  attn_kernel<<<BB*LL*HH, 384, SH_ATTN, stream>>>(QKVFU, PBUF, mask, OBUF, 1);
  outproj_kernel<<<MM/128, 256, SH_PROJ, stream>>>(OBUF, WO_B, aout_bo, mask, PR1, PR2, PR2BF, 1);

  // ---- transition convs ----
  conv1_kernel<<<BB*LL*3, 256, SH_C1, stream>>>(PR2BF, CW1, c1b, T1);
  conv2_kernel<<<BB*LL*3, 256, SH_C2, stream>>>(T1, CW2, c2b, PR2, mask, out);
}
